// GraphTransformerLayer_91276644974953
// MI455X (gfx1250) — matrix operand order repaired and checked by reading, not compiled, not run on hardware
//
#include <hip/hip_runtime.h>
#include <hip/hip_bf16.h>
#include <math.h>

typedef __attribute__((ext_vector_type(16))) _Float16 v16h;
typedef __attribute__((ext_vector_type(8)))  _Float16 h8;
typedef __attribute__((ext_vector_type(4)))  _Float16 h4;
typedef __attribute__((ext_vector_type(8)))  float    v8f;

#define TM 128      // block tile M
#define TN 64       // block tile N
#define TK 32       // K step
#define SA_STR 40   // halfs; 80B row stride -> every b128 access 16B-aligned
#define SB_STR 40

__device__ __forceinline__ float gelu_exact(float v) {
    return 0.5f * v * (1.0f + erff(v * 0.70710678118654752f));
}

// Order-preserving float <-> uint encoding so float max == integer atomicMax.
__device__ __forceinline__ unsigned int encF(float f) {
    unsigned int u = __float_as_uint(f);
    return (u & 0x80000000u) ? ~u : (u | 0x80000000u);
}
__device__ __forceinline__ float decF(unsigned int u) {
    return (u & 0x80000000u) ? __uint_as_float(u & 0x7FFFFFFFu)
                             : __uint_as_float(~u);
}

// K-column permutation so each lane's A fragment is 16 contiguous halfs.
// Lane kh=0 needs K = 0..7,16..23 ; kh=1 needs K = 8..15,24..31.
__device__ __forceinline__ int permK(int k) {
    return (k & 7) | ((k & 8) << 1) | ((k & 16) >> 1);
}

__device__ __forceinline__ v16h frag16(const _Float16* base, int kh) {
    const h8* p = (const h8*)base;
    h8 lo = p[kh * 2 + 0];
    h8 hi = p[kh * 2 + 1];
    return __builtin_shufflevector(lo, hi, 0,1,2,3,4,5,6,7,8,9,10,11,12,13,14,15);
}

// ---------------------------------------------------------------------------
// fp32-in / f16-WMMA / fp32-out GEMM:  C = act(A[MxK] * B[KxN] + bias)
// Block: 256 threads = 8 waves arranged 4(M) x 2(N); each wave computes a
// 32x32 tile (2x2 WMMA register blocking, 4 WMMAs/K-step) => block 128x64.
// K multiple of 32, Nout multiple of 64.  act: 0 = none, 1 = exact GELU.
// Out-of-range M rows are CLAMPED on load (never stored) to keep all global
// loads/prefetches unpredicated.
// ---------------------------------------------------------------------------
__global__ __launch_bounds__(256)
void gemm_f16wmma(const float* __restrict__ A, const float* __restrict__ B,
                  const float* __restrict__ bias, float* __restrict__ C,
                  int M, int K, int Nout, int lda, int ldb, int ldc, int act)
{
    __shared__ alignas(16) _Float16 sA[TM * SA_STR];   // [row][permK]  10240 B
    __shared__ alignas(16) _Float16 sBt[TN * SB_STR];  // [n][k] transp. 5120 B

    const int tid  = threadIdx.x;
    const int wave = tid >> 5;
    const int lane = tid & 31;
    const int wm   = wave >> 1;   // 0..3 -> M offset 32*wm
    const int wn   = wave & 1;    // 0..1 -> N offset 32*wn
    const int blockM = blockIdx.x * TM;
    const int blockN = blockIdx.y * TN;

    const int m  = lane & 15;
    const int kh = lane >> 4;

    v8f acc00 = {}, acc01 = {}, acc10 = {}, acc11 = {};

    for (int k0 = 0; k0 < K; k0 += TK) {
        const bool more = (k0 + TK) < K;   // uniform
        // ---- Stage A tile: 128x32 fp32 -> f16, permuted K, float4 loads ----
#pragma unroll
        for (int t = 0; t < 4; ++t) {
            int i4 = tid + t * 256;            // 1024 float4s total
            int r  = i4 >> 3;                  // row 0..127
            int cg = (i4 & 7) * 4;             // K group base
            int gr = blockM + r; if (gr > M - 1) gr = M - 1;   // clamp, no divergence
            const float* ap = A + (size_t)gr * lda + (k0 + cg);
            float4 fv = *(const float4*)ap;
            if (more) __builtin_prefetch(ap + TK, 0, 0);
            h4 hv; hv[0] = (_Float16)fv.x; hv[1] = (_Float16)fv.y;
                   hv[2] = (_Float16)fv.z; hv[3] = (_Float16)fv.w;
            *(h4*)(sA + r * SA_STR + permK(cg)) = hv;   // 8B-aligned b64 store
        }
        // ---- Stage B tile: 32x64 fp32 -> f16, transposed to [n][k] ----
#pragma unroll
        for (int t = 0; t < 2; ++t) {
            int i4 = tid + t * 256;            // 512 float4s total
            int r  = i4 >> 4;                  // k row 0..31
            int cg = (i4 & 15) * 4;            // n group base
            const float* bp = B + (size_t)(k0 + r) * ldb + (blockN + cg);
            float4 fv = *(const float4*)bp;
            if (more) __builtin_prefetch(bp + (size_t)TK * ldb, 0, 0);
            sBt[(cg + 0) * SB_STR + permK(r)] = (_Float16)fv.x;
            sBt[(cg + 1) * SB_STR + permK(r)] = (_Float16)fv.y;
            sBt[(cg + 2) * SB_STR + permK(r)] = (_Float16)fv.z;
            sBt[(cg + 3) * SB_STR + permK(r)] = (_Float16)fv.w;
        }
        __syncthreads();

        // ---- Fragments: 2x ds_load_b128 each (16 contiguous halfs) ----
        v16h af0 = frag16(sA  + (wm * 32 +      m) * SA_STR, kh);
        v16h af1 = frag16(sA  + (wm * 32 + 16 + m) * SA_STR, kh);
        v16h bf0 = frag16(sBt + (wn * 32 +      m) * SB_STR, kh);
        v16h bf1 = frag16(sBt + (wn * 32 + 16 + m) * SB_STR, kh);

        acc00 = __builtin_amdgcn_wmma_f32_16x16x32_f16(false, af0, false, bf0, (short)0, acc00, false, false);
        acc01 = __builtin_amdgcn_wmma_f32_16x16x32_f16(false, af0, false, bf1, (short)0, acc01, false, false);
        acc10 = __builtin_amdgcn_wmma_f32_16x16x32_f16(false, af1, false, bf0, (short)0, acc10, false, false);
        acc11 = __builtin_amdgcn_wmma_f32_16x16x32_f16(false, af1, false, bf1, (short)0, acc11, false, false);
        __syncthreads();
    }

    // Epilogue: D layout -> VGPR r holds M = r + 8*(lane>=16), N = lane&15.
    const int col0 = blockN + wn * 32 + m;
    const int col1 = col0 + 16;
    const float bi0 = bias[col0];
    const float bi1 = bias[col1];
#pragma unroll
    for (int r = 0; r < 8; ++r) {
        int row0 = blockM + wm * 32 + r + 8 * kh;
        int row1 = row0 + 16;
        if (row0 < M) {
            float v0 = acc00[r] + bi0, v1 = acc01[r] + bi1;
            if (act == 1) { v0 = gelu_exact(v0); v1 = gelu_exact(v1); }
            C[(size_t)row0 * ldc + col0] = v0;
            C[(size_t)row0 * ldc + col1] = v1;
        }
        if (row1 < M) {
            float v0 = acc10[r] + bi0, v1 = acc11[r] + bi1;
            if (act == 1) { v0 = gelu_exact(v0); v1 = gelu_exact(v1); }
            C[(size_t)row1 * ldc + col0] = v0;
            C[(size_t)row1 * ldc + col1] = v1;
        }
    }
}

// ---------------------------------------------------------------------------
// Init: zero aggregation + softmax sums, set encoded segment-max to enc(-inf).
// ---------------------------------------------------------------------------
__global__ void init_state(float* agg, float* ssum, unsigned int* mEnc, int N)
{
    int i = blockIdx.x * blockDim.x + threadIdx.x;
    if (i < N * 256) agg[i] = 0.0f;
    if (i < N * 4) { ssum[i] = 0.0f; mEnc[i] = 0x007FFFFFu; /* enc(-inf) */ }
}

// ---------------------------------------------------------------------------
// Pass 1: wave per edge. alpha[e][h] = q[dst]·(k[src]+e) / sqrt(64).
// Lane l covers channels [8l, 8l+8) via two float4 loads; head = l>>3.
// ---------------------------------------------------------------------------
__global__ __launch_bounds__(256)
void attn_alpha(const float* __restrict__ qkvs, const float* __restrict__ eproj,
                const int* __restrict__ src, const int* __restrict__ dst,
                float* __restrict__ alpha, unsigned int* __restrict__ mEnc, int E)
{
    int e = blockIdx.x * 8 + (threadIdx.x >> 5);
    if (e >= E) return;
    int lane = threadIdx.x & 31;
    int s = src[e], d = dst[e];
    const float4* q4 = (const float4*)(qkvs + (size_t)d * 1024);        // q
    const float4* k4 = (const float4*)(qkvs + (size_t)s * 1024 + 256);  // k
    const float4* e4 = (const float4*)(eproj + (size_t)e * 256);

    float p = 0.0f;
#pragma unroll
    for (int t = 0; t < 2; ++t) {
        float4 qv = q4[lane * 2 + t];
        float4 kv = k4[lane * 2 + t];
        float4 ev = e4[lane * 2 + t];
        p += qv.x * (kv.x + ev.x) + qv.y * (kv.y + ev.y)
           + qv.z * (kv.z + ev.z) + qv.w * (kv.w + ev.w);
    }
    p += __shfl_xor(p, 1, 32);
    p += __shfl_xor(p, 2, 32);
    p += __shfl_xor(p, 4, 32);
    if ((lane & 7) == 0) {
        int h = lane >> 3;
        float a = p * 0.125f;                          // 1/sqrt(C), C=64
        alpha[(size_t)e * 4 + h] = a;
        atomicMax(&mEnc[(size_t)d * 4 + h], encF(a));
    }
}

// ---------------------------------------------------------------------------
// Pass 2: thread per (edge, head). a = exp(alpha - m[dst]); s[dst] += a.
// ---------------------------------------------------------------------------
__global__ void attn_expsum(float* __restrict__ alpha, const int* __restrict__ dst,
                            const unsigned int* __restrict__ mEnc,
                            float* __restrict__ ssum, int E)
{
    int i = blockIdx.x * blockDim.x + threadIdx.x;
    if (i >= E * 4) return;
    int e = i >> 2, h = i & 3;
    int d = dst[e];
    float m = decF(mEnc[(size_t)d * 4 + h]);
    float a = __expf(alpha[i] - m);
    alpha[i] = a;
    atomicAdd(&ssum[(size_t)d * 4 + h], a);
}

// ---------------------------------------------------------------------------
// Pass 3: wave per edge. agg[dst] += (v[src] + e) * a / (s + eps).
// ---------------------------------------------------------------------------
__global__ __launch_bounds__(256)
void attn_scatter(const float* __restrict__ qkvs, const float* __restrict__ eproj,
                  const float* __restrict__ alpha, const float* __restrict__ ssum,
                  const int* __restrict__ src, const int* __restrict__ dst,
                  float* __restrict__ agg, int E)
{
    int e = blockIdx.x * 8 + (threadIdx.x >> 5);
    if (e >= E) return;
    int lane = threadIdx.x & 31;
    int s = src[e], d = dst[e];
    int h = lane >> 3;
    float w = alpha[(size_t)e * 4 + h] / (ssum[(size_t)d * 4 + h] + 1e-16f);
    const float4* v4 = (const float4*)(qkvs + (size_t)s * 1024 + 512);  // v
    const float4* e4 = (const float4*)(eproj + (size_t)e * 256);
    float* ag = agg + (size_t)d * 256;
#pragma unroll
    for (int t = 0; t < 2; ++t) {
        float4 vv = v4[lane * 2 + t];
        float4 ev = e4[lane * 2 + t];
        int ch = lane * 8 + t * 4;
        atomicAdd(ag + ch + 0, (vv.x + ev.x) * w);
        atomicAdd(ag + ch + 1, (vv.y + ev.y) * w);
        atomicAdd(ag + ch + 2, (vv.z + ev.z) * w);
        atomicAdd(ag + ch + 3, (vv.w + ev.w) * w);
    }
}

// ---------------------------------------------------------------------------
// Wave per node: beta gating + residual + LayerNorm1 -> x1.
// ---------------------------------------------------------------------------
__global__ __launch_bounds__(256)
void beta_ln1(const float* __restrict__ x, const float* __restrict__ qkvs,
              const float* __restrict__ agg, const float* __restrict__ Wbeta,
              const float* __restrict__ g, const float* __restrict__ b,
              float* __restrict__ x1, int N)
{
    int n = blockIdx.x * 8 + (threadIdx.x >> 5);
    if (n >= N) return;
    int lane = threadIdx.x & 31;

    const float4* o4  = (const float4*)(agg + (size_t)n * 256);
    const float4* xr4 = (const float4*)(qkvs + (size_t)n * 1024 + 768);  // skip
    const float4* xv4 = (const float4*)(x + (size_t)n * 256);
    const float4* w0  = (const float4*)(Wbeta);
    const float4* w1  = (const float4*)(Wbeta + 256);
    const float4* w2  = (const float4*)(Wbeta + 512);

    float o[8], xr[8], xv[8];
    float t = 0.0f;
#pragma unroll
    for (int q = 0; q < 2; ++q) {
        int i4 = lane * 2 + q;
        float4 ov = o4[i4], rv = xr4[i4], vv = xv4[i4];
        float4 a0 = w0[i4], a1 = w1[i4], a2 = w2[i4];
        o[q*4+0] = ov.x; o[q*4+1] = ov.y; o[q*4+2] = ov.z; o[q*4+3] = ov.w;
        xr[q*4+0] = rv.x; xr[q*4+1] = rv.y; xr[q*4+2] = rv.z; xr[q*4+3] = rv.w;
        xv[q*4+0] = vv.x; xv[q*4+1] = vv.y; xv[q*4+2] = vv.z; xv[q*4+3] = vv.w;
        t += ov.x*a0.x + ov.y*a0.y + ov.z*a0.z + ov.w*a0.w;
        t += rv.x*a1.x + rv.y*a1.y + rv.z*a1.z + rv.w*a1.w;
        t += (ov.x-rv.x)*a2.x + (ov.y-rv.y)*a2.y + (ov.z-rv.z)*a2.z + (ov.w-rv.w)*a2.w;
    }
#pragma unroll
    for (int mk = 1; mk < 32; mk <<= 1) t += __shfl_xor(t, mk, 32);
    float beta = 1.0f / (1.0f + __expf(-t));

    float sum = 0.0f;
#pragma unroll
    for (int j = 0; j < 8; ++j) {
        float y = beta * xr[j] + (1.0f - beta) * o[j] + xv[j];   // + residual
        o[j] = y;
        sum += y;
    }
#pragma unroll
    for (int mk = 1; mk < 32; mk <<= 1) sum += __shfl_xor(sum, mk, 32);
    float mu = sum * (1.0f / 256.0f);

    float sq = 0.0f;
#pragma unroll
    for (int j = 0; j < 8; ++j) { float dv = o[j] - mu; sq += dv * dv; }
#pragma unroll
    for (int mk = 1; mk < 32; mk <<= 1) sq += __shfl_xor(sq, mk, 32);
    float rstd = rsqrtf(sq * (1.0f / 256.0f) + 1e-5f);

#pragma unroll
    for (int j = 0; j < 8; ++j) {
        int ch = lane * 8 + j;
        x1[(size_t)n * 256 + ch] = (o[j] - mu) * rstd * g[ch] + b[ch];
    }
}

// ---------------------------------------------------------------------------
// Wave per node: out = LayerNorm2(x1 + h2).
// ---------------------------------------------------------------------------
__global__ __launch_bounds__(256)
void residual_ln2(const float* __restrict__ x1, const float* __restrict__ h2,
                  const float* __restrict__ g, const float* __restrict__ b,
                  float* __restrict__ out, int N)
{
    int n = blockIdx.x * 8 + (threadIdx.x >> 5);
    if (n >= N) return;
    int lane = threadIdx.x & 31;

    const float4* a4 = (const float4*)(x1 + (size_t)n * 256);
    const float4* b4 = (const float4*)(h2 + (size_t)n * 256);

    float y[8];
    float sum = 0.0f;
#pragma unroll
    for (int q = 0; q < 2; ++q) {
        float4 av = a4[lane * 2 + q], bv = b4[lane * 2 + q];
        y[q*4+0] = av.x + bv.x; y[q*4+1] = av.y + bv.y;
        y[q*4+2] = av.z + bv.z; y[q*4+3] = av.w + bv.w;
        sum += y[q*4+0] + y[q*4+1] + y[q*4+2] + y[q*4+3];
    }
#pragma unroll
    for (int mk = 1; mk < 32; mk <<= 1) sum += __shfl_xor(sum, mk, 32);
    float mu = sum * (1.0f / 256.0f);

    float sq = 0.0f;
#pragma unroll
    for (int j = 0; j < 8; ++j) { float dv = y[j] - mu; sq += dv * dv; }
#pragma unroll
    for (int mk = 1; mk < 32; mk <<= 1) sq += __shfl_xor(sq, mk, 32);
    float rstd = rsqrtf(sq * (1.0f / 256.0f) + 1e-5f);

#pragma unroll
    for (int j = 0; j < 8; ++j) {
        int ch = lane * 8 + j;
        out[(size_t)n * 256 + ch] = (y[j] - mu) * rstd * g[ch] + b[ch];
    }
}

// ---------------------------------------------------------------------------
extern "C" void kernel_launch(void* const* d_in, const int* in_sizes, int n_in,
                              void* d_out, int out_size, void* d_ws, size_t ws_size,
                              hipStream_t stream)
{
    (void)n_in; (void)out_size; (void)ws_size;

    const float* x         = (const float*)d_in[0];
    const int*   edge_idx  = (const int*)  d_in[1];
    const float* edge_attr = (const float*)d_in[2];
    const float* Wq    = (const float*)d_in[3];
    const float* bq    = (const float*)d_in[4];
    const float* Wk    = (const float*)d_in[5];
    const float* bk    = (const float*)d_in[6];
    const float* Wv    = (const float*)d_in[7];
    const float* bv    = (const float*)d_in[8];
    const float* We    = (const float*)d_in[9];
    const float* be    = (const float*)d_in[10];
    const float* Wskip = (const float*)d_in[11];
    const float* bskip = (const float*)d_in[12];
    const float* Wbeta = (const float*)d_in[13];
    const float* ln1_g = (const float*)d_in[14];
    const float* ln1_b = (const float*)d_in[15];
    const float* W1    = (const float*)d_in[16];
    const float* b1    = (const float*)d_in[17];
    const float* W2    = (const float*)d_in[18];
    const float* b2    = (const float*)d_in[19];
    const float* ln2_g = (const float*)d_in[20];
    const float* ln2_b = (const float*)d_in[21];

    const int N = in_sizes[0] / 256;   // 20000
    const int E = in_sizes[2] / 256;   // 200000
    const int* src = edge_idx;         // edge_index[0]
    const int* dst = edge_idx + E;     // edge_index[1]

    // Workspace carve-up (fp32 words).
    float* ws = (float*)d_ws;
    size_t off = 0;
    float* qkvs  = ws + off; off += (size_t)N * 1024;  // [q|k|v|skip]
    float* eproj = ws + off; off += (size_t)E * 256;
    float* alpha = ws + off; off += (size_t)E * 4;
    unsigned int* mEnc = (unsigned int*)(ws + off); off += (size_t)N * 4;
    float* ssum  = ws + off; off += (size_t)N * 4;
    float* agg   = ws + off; off += (size_t)N * 256;   // reused for h2 later
    float* x1    = ws + off; off += (size_t)N * 256;
    float* h1    = ws + off; off += (size_t)N * 512;
    float* h2    = agg;                                // agg dead after beta_ln1

    dim3 blk(256);

    init_state<<<(N * 256 + 255) / 256, blk, 0, stream>>>(agg, ssum, mEnc, N);

    // Node projections -> fused qkvs buffer (ldc = 1024, column offsets).
    dim3 gN((N + TM - 1) / TM, 256 / TN);
    gemm_f16wmma<<<gN, blk, 0, stream>>>(x, Wq,    bq,    qkvs +   0, N, 256, 256, 256, 256, 1024, 0);
    gemm_f16wmma<<<gN, blk, 0, stream>>>(x, Wk,    bk,    qkvs + 256, N, 256, 256, 256, 256, 1024, 0);
    gemm_f16wmma<<<gN, blk, 0, stream>>>(x, Wv,    bv,    qkvs + 512, N, 256, 256, 256, 256, 1024, 0);
    gemm_f16wmma<<<gN, blk, 0, stream>>>(x, Wskip, bskip, qkvs + 768, N, 256, 256, 256, 256, 1024, 0);

    // Edge-feature projection.
    dim3 gE((E + TM - 1) / TM, 256 / TN);
    gemm_f16wmma<<<gE, blk, 0, stream>>>(edge_attr, We, be, eproj, E, 256, 256, 256, 256, 256, 0);

    // Segment softmax attention (3 passes) + scatter aggregation.
    attn_alpha  <<<(E + 7) / 8,         blk, 0, stream>>>(qkvs, eproj, src, dst, alpha, mEnc, E);
    attn_expsum <<<(E * 4 + 255) / 256, blk, 0, stream>>>(alpha, dst, mEnc, ssum, E);
    attn_scatter<<<(E + 7) / 8,         blk, 0, stream>>>(qkvs, eproj, alpha, ssum, src, dst, agg, E);

    // Beta gating + residual + LN1.
    beta_ln1<<<(N + 7) / 8, blk, 0, stream>>>(x, qkvs, agg, Wbeta, ln1_g, ln1_b, x1, N);

    // FFN: GELU(x1@W1 + b1) @ W2 + b2.
    dim3 g1((N + TM - 1) / TM, 512 / TN);
    gemm_f16wmma<<<g1, blk, 0, stream>>>(x1, W1, b1, h1, N, 256, 512, 256, 512, 512, 1);
    dim3 g2((N + TM - 1) / TM, 256 / TN);
    gemm_f16wmma<<<g2, blk, 0, stream>>>(h1, W2, b2, h2, N, 512, 256, 512, 256, 256, 0);

    // Residual + LN2 -> output.
    residual_ln2<<<(N + 7) / 8, blk, 0, stream>>>(x1, h2, ln2_g, ln2_b, (float*)d_out, N);
}
